// MultiHeadAttention_5652176961950
// MI455X (gfx1250) — matrix operand order repaired and checked by reading, not compiled, not run on hardware
//
#include <hip/hip_runtime.h>
#include <hip/hip_bf16.h>

// MI455X / gfx1250 multi-head attention forward.
// Reference shapes: B=8, S=1024, D=512, H=8, DK=64.
// Strategy: f16-operand / f32-accumulate WMMA (v_wmma_f32_16x16x32_f16) for
// all four GEMMs; fused softmax+PV keeps probabilities in LDS (320KB/WGP)
// so the only big HBM traffic is the mandatory 256MB attn output write.
// Data movement uses the CDNA5 paths: TDM tensor_load_to_lds for weight
// tiles (TENSORcnt) and global_load_async_to_lds_b128 for Q strips (ASYNCcnt).

#define B_  8
#define S_  1024
#define D_  512
#define H_  8
#define DK_ 64

typedef __attribute__((ext_vector_type(16))) _Float16 v16h;
typedef __attribute__((ext_vector_type(8)))  float    v8f;
typedef __attribute__((ext_vector_type(4)))  unsigned v4u;
typedef __attribute__((ext_vector_type(8)))  unsigned v8u;

__device__ __forceinline__ v8f wmma_f16(const v16h& a, const v16h& b, const v8f& c) {
  // D = A(16x32 f16) * B(32x16 f16) + C(16x16 f32)
  return __builtin_amdgcn_wmma_f32_16x16x32_f16(
      /*neg_a=*/false, a, /*neg_b=*/false, b,
      /*c_mod=*/(short)0, c, /*reuse_a=*/false, /*reuse_b=*/false);
}

// A-fragment K index for 16-bit 16x32 A: lanes 0-15 and
// 16-31 both hold M=lane&15; element j maps to K = 16*g + j (matches B).
__device__ __forceinline__ int a_kidx(int j, int g) {
  return 16 * g + j;
}
// B-fragment: lane group g holds K = 16*g + j for N = lane&15 (this code's contiguous order,
// same as A).

// ---------------------------------------------------------------------------
// Kernel 1: Q/K/V projections -> f16 operand copies (+bias), fp32 accum.
// job y: 0..7 = Q head, 8..15 = K head, 16 = shared V projection.
// Block: 256 threads (8 waves), each wave owns a 16-row M strip (M tile 128).
// Weight K-chunks [32][64] fp32 are DMA'd into LDS by the Tensor Data Mover
// with pad_enable giving a 65-DWORD LDS row pitch (bank-conflict padding).
// ---------------------------------------------------------------------------
__global__ __launch_bounds__(256) void proj_kernel(
    const float* __restrict__ q, const float* __restrict__ k,
    const float* __restrict__ v,
    const float* __restrict__ Wq, const float* __restrict__ bq,
    const float* __restrict__ Wk, const float* __restrict__ bk,
    const float* __restrict__ Wv, const float* __restrict__ bv,
    _Float16* __restrict__ qs, _Float16* __restrict__ ks,
    _Float16* __restrict__ vs) {
  __shared__ float lw[32][DK_ + 1];   // TDM-padded weight K-chunk (row pitch 65)

  const int job = blockIdx.y;
  const float* X; const float* W; const float* bias; _Float16* dst; int head;
  if (job < 8)        { X = q; W = Wq + (size_t)job * D_ * DK_;       bias = bq + job * DK_;       dst = qs; head = job; }
  else if (job < 16)  { X = k; W = Wk + (size_t)(job - 8) * D_ * DK_; bias = bk + (job - 8) * DK_; dst = ks; head = job - 8; }
  else                { X = v; W = Wv;                                bias = bv;                   dst = vs; head = 0; }

  const int tid  = threadIdx.x;
  const int wave = tid >> 5, lane = tid & 31;
  const int g = lane >> 4, l16 = lane & 15;
  const int m0 = blockIdx.x * 128 + wave * 16;
  const float* xrow = X + (size_t)(m0 + l16) * D_;

  v8f acc[4] = {};

  for (int kk = 0; kk < D_; kk += 32) {
    // --- TDM: one wave issues tensor_load_to_lds for W[kk..kk+31][0..63] ---
    if (wave == 0) {
      const unsigned lds_base = (unsigned)(size_t)(&lw[0][0]);
      const unsigned long long ga =
          (unsigned long long)(const void*)(W + (size_t)kk * DK_);
      // D# group 0: count=1, lds_addr, global_addr[56:0], type=2 (image).
      v4u g0;
      g0[0] = 1u;
      g0[1] = lds_base;
      g0[2] = (unsigned)ga;
      g0[3] = ((unsigned)(ga >> 32) & 0x1FFFFFFu) | (2u << 30);
      // D# group 1: data_size=4B, pad every 64 DWORDs by 1 DWORD,
      // tensor 64x32, tile 64x32, row stride 64 elements.
      v8u g1;
      g1[0] = (2u << 16) | (1u << 20) | (5u << 22);   // size|pad_en|pad_int=64dw
      g1[1] = (64u << 16);                            // tensor_dim0 = 64
      g1[2] = (32u << 16);                            // tensor_dim1 = 32
      g1[3] = (64u << 16);                            // tile_dim0 = 64
      g1[4] = 32u;                                    // tile_dim1=32, tile_dim2=0
      g1[5] = 64u;                                    // tensor_dim0_stride = 64
      g1[6] = 0u;
      g1[7] = 0u;
      // D# groups 2/3: unused upper dims set to 1, tiles 0 (=unused).
      v4u g2; g2[0] = 1u; g2[1] = 1u; g2[2] = 0u; g2[3] = 0u;
      v4u g3; g3[0] = 0u; g3[1] = (1u << 16); g3[2] = 0u; g3[3] = 0u;
      asm volatile("tensor_load_to_lds %0, %1, %2, %3"
                   :: "s"(g0), "s"(g1), "s"(g2), "s"(g3)
                   : "memory");
      __builtin_amdgcn_s_wait_tensorcnt(0);
    }
    // Hint the next activation chunk toward the caches (global_prefetch_b8).
    __builtin_prefetch(xrow + kk + 32, 0, 1);
    __syncthreads();

    // A fragment: one contiguous 16-float run (K = 16*g + j) of this lane's activation row.
    v16h a;
    const float* r1 = xrow + kk + 16 * g;
    const float* r2 = xrow + kk + 16 * g + 8;
#pragma unroll
    for (int j = 0; j < 8; j++) a[j]     = (_Float16)r1[j];
#pragma unroll
    for (int j = 0; j < 8; j++) a[j + 8] = (_Float16)r2[j];

#pragma unroll
    for (int t = 0; t < 4; t++) {
      v16h bf;
#pragma unroll
      for (int j = 0; j < 16; j++)
        bf[j] = (_Float16)lw[16 * g + j][t * 16 + l16];
      acc[t] = wmma_f16(a, bf, acc[t]);
    }
    __syncthreads();
  }

  // Bias + store f16 operands. C/D layout: VGPR r -> M = r + 8*g, N = lane&15.
#pragma unroll
  for (int t = 0; t < 4; t++) {
    const int n = t * 16 + l16;
    const float bn = bias[n];
#pragma unroll
    for (int r = 0; r < 8; r++) {
      const int mrow = m0 + r + 8 * g;
      const int bb = mrow >> 10, ss = mrow & (S_ - 1);
      const float val = acc[t][r] + bn;
      size_t idx;
      if (job < 16) idx = (((size_t)(bb * H_ + head) * S_ + ss) * DK_ + n);
      else          idx = (((size_t)bb * S_ + ss) * DK_ + n);
      dst[idx] = (_Float16)val;
    }
  }
}

// ---------------------------------------------------------------------------
// Kernel 2: fused QK^T/sqrt(DK) -> softmax -> attn write -> PV.
// Block: 256 threads (8 waves) owns a 16-query strip of one (b,h).
// The Q strip (16x64 f16 = 2KB) is staged to LDS with async-to-LDS b128.
// Scores strip [16][1024] lives in LDS; waves split the 1024-key dimension.
// ---------------------------------------------------------------------------
__global__ __launch_bounds__(256) void attn_kernel(
    const _Float16* __restrict__ qs, const _Float16* __restrict__ ks,
    const _Float16* __restrict__ vs, float* __restrict__ attn_out,
    float* __restrict__ heads) {
  __shared__ float sc[16][S_ + 1];        // 65.6 KB probability strip
  __shared__ float hacc[8][16][DK_];      // 32 KB per-wave PV partials
  __shared__ _Float16 qtile[16][DK_];     // 2 KB async-staged Q strip

  const int bh = blockIdx.y;
  const int b = bh >> 3, h = bh & 7;
  const int strip = blockIdx.x;           // 16-query strip index
  const int tid = threadIdx.x, wave = tid >> 5, lane = tid & 31;
  const int g = lane >> 4, l16 = lane & 15;

  const _Float16* qbase = qs + ((size_t)(b * H_ + h) * S_ + strip * 16) * DK_;
  const _Float16* khead = ks + (size_t)(b * H_ + h) * S_ * DK_;

  // ---- async copy Q strip (2048B = 128 lanes x 16B) into LDS ----
  if (tid < 128) {                        // waves 0-3, wave-uniform branch
    const unsigned ldsoff = (unsigned)(size_t)(&qtile[0][0]) + tid * 16;
    const unsigned goff   = tid * 16;     // byte offset from qbase
    asm volatile("global_load_async_to_lds_b128 %0, %1, %2"
                 :: "v"(ldsoff), "v"(goff),
                    "s"((unsigned long long)(const void*)qbase)
                 : "memory");
  }
  asm volatile("s_wait_asynccnt 0x0" ::: "memory");
  __syncthreads();

  // Q fragments (two K=32 chunks covering DK=64), shared by every n-tile.
  v16h a0, a1;
#pragma unroll
  for (int j = 0; j < 16; j++) {
    const int kd = a_kidx(j, g);
    a0[j] = qtile[l16][kd];
    a1[j] = qtile[l16][32 + kd];
  }

  // ---- scores: wave `w` computes key n-tiles w, w+8, ... ----
  for (int nt = wave; nt < S_ / 16; nt += 8) {
    const _Float16* kr = khead + (size_t)(nt * 16 + l16) * DK_;
    v16h b0, b1;
#pragma unroll
    for (int j = 0; j < 16; j++) {
      b0[j] = kr[16 * g + j];        // B^T gather: contiguous dk run per lane
      b1[j] = kr[32 + 16 * g + j];
    }
    v8f c = {};
    c = wmma_f16(a0, b0, c);
    c = wmma_f16(a1, b1, c);
#pragma unroll
    for (int r = 0; r < 8; r++)
      sc[r + 8 * g][nt * 16 + l16] = c[r] * 0.125f;   // 1/sqrt(64)
  }
  __syncthreads();

  // ---- softmax: 16 lanes per query row, wave32-native shfl_xor trees ----
  {
    const int r = tid >> 4, sub = tid & 15;
    float mx = -3.0e38f;
    for (int i = 0; i < 64; i++) mx = fmaxf(mx, sc[r][sub + 16 * i]);
#pragma unroll
    for (int off = 1; off < 16; off <<= 1) mx = fmaxf(mx, __shfl_xor(mx, off, 32));

    float sum = 0.f;
    for (int i = 0; i < 64; i++) {
      const float e = __expf(sc[r][sub + 16 * i] - mx);
      sc[r][sub + 16 * i] = e;
      sum += e;
    }
#pragma unroll
    for (int off = 1; off < 16; off <<= 1) sum += __shfl_xor(sum, off, 32);
    const float inv = 1.f / sum;

    // attn output layout is [B, S, H, S] (reference transposes to (0,2,1,3)).
    const size_t obase = (((size_t)b * S_ + strip * 16 + r) * H_ + h) * S_;
    for (int i = 0; i < 64; i++) {
      const float p = sc[r][sub + 16 * i] * inv;
      sc[r][sub + 16 * i] = p;
      attn_out[obase + sub + 16 * i] = p;
    }
  }
  __syncthreads();

  // ---- PV: each wave reduces 128 consecutive keys, partials in LDS ----
  const _Float16* vhead = vs + (size_t)b * S_ * DK_;
  v8f acc[4] = {};
  for (int ck = 0; ck < 4; ck++) {
    const int kk = wave * 128 + ck * 32;
    v16h a;
#pragma unroll
    for (int j = 0; j < 16; j++)
      a[j] = (_Float16)sc[l16][kk + a_kidx(j, g)];
#pragma unroll
    for (int t = 0; t < 4; t++) {
      v16h bf;
#pragma unroll
      for (int j = 0; j < 16; j++)
        bf[j] = vhead[(size_t)(kk + 16 * g + j) * DK_ + t * 16 + l16];
      acc[t] = wmma_f16(a, bf, acc[t]);
    }
  }
#pragma unroll
  for (int t = 0; t < 4; t++)
#pragma unroll
    for (int r = 0; r < 8; r++)
      hacc[wave][r + 8 * g][t * 16 + l16] = acc[t][r];
  __syncthreads();

  // Deterministic cross-wave reduction; heads layout [B,H,S,DK] fp32.
  for (int e = tid; e < 16 * DK_; e += 256) {
    const int r = e >> 6, n = e & 63;
    float s = 0.f;
#pragma unroll
    for (int w2 = 0; w2 < 8; w2++) s += hacc[w2][r][n];
    heads[((size_t)(b * H_ + h) * S_ + strip * 16 + r) * DK_ + n] = s;
  }
}

// ---------------------------------------------------------------------------
// Kernel 3: outputs = mean_h(heads) @ Wh.  GEMM [8192,64] x [64,512].
// Block handles a 128-row M tile; head-mean staged in LDS.
// ---------------------------------------------------------------------------
__global__ __launch_bounds__(256) void out_kernel(
    const float* __restrict__ heads, const float* __restrict__ Wh,
    float* __restrict__ out) {
  __shared__ float hm[128][DK_ + 1];   // 33 KB mean-of-heads tile

  const int tid = threadIdx.x, wave = tid >> 5, lane = tid & 31;
  const int g = lane >> 4, l16 = lane & 15;
  const int m0 = blockIdx.x * 128;

  for (int e = tid; e < 128 * DK_; e += 256) {
    const int r = e >> 6, n = e & 63;
    const int mrow = m0 + r, bb = mrow >> 10, ss = mrow & (S_ - 1);
    float acc = 0.f;
#pragma unroll
    for (int hh = 0; hh < H_; hh++)
      acc += heads[((size_t)(bb * H_ + hh) * S_ + ss) * DK_ + n];
    hm[r][n] = acc * 0.125f;   // 1/H
  }
  __syncthreads();

  const int mw = wave * 16;
  v16h a0, a1;
#pragma unroll
  for (int j = 0; j < 16; j++) {
    const int kd = a_kidx(j, g);
    a0[j] = (_Float16)hm[mw + l16][kd];
    a1[j] = (_Float16)hm[mw + l16][32 + kd];
  }

  for (int nt = 0; nt < D_ / 16; nt++) {
    const int n = nt * 16 + l16;
    v16h b0, b1;
#pragma unroll
    for (int j = 0; j < 16; j++) {
      b0[j] = (_Float16)Wh[(size_t)(16 * g + j) * D_ + n];
      b1[j] = (_Float16)Wh[(size_t)(32 + 16 * g + j) * D_ + n];
    }
    v8f c = {};
    c = wmma_f16(a0, b0, c);
    c = wmma_f16(a1, b1, c);
#pragma unroll
    for (int r = 0; r < 8; r++) {
      const int mrow = m0 + mw + r + 8 * g;
      out[(size_t)mrow * D_ + n] = c[r];
    }
  }
}

// ---------------------------------------------------------------------------
extern "C" void kernel_launch(void* const* d_in, const int* in_sizes, int n_in,
                              void* d_out, int out_size, void* d_ws, size_t ws_size,
                              hipStream_t stream) {
  (void)in_sizes; (void)n_in; (void)out_size; (void)ws_size;
  const float* q  = (const float*)d_in[0];
  const float* k  = (const float*)d_in[1];
  const float* v  = (const float*)d_in[2];
  const float* Wq = (const float*)d_in[3];
  const float* bq = (const float*)d_in[4];
  const float* Wk = (const float*)d_in[5];
  const float* bk = (const float*)d_in[6];
  const float* Wv = (const float*)d_in[7];
  const float* bv = (const float*)d_in[8];
  const float* Wh = (const float*)d_in[9];

  float* out_main = (float*)d_out;                           // [B,S,D]
  float* out_attn = out_main + (size_t)B_ * S_ * D_;         // [B,S,H,S]

  char* ws = (char*)d_ws;
  _Float16* qs = (_Float16*)ws; ws += (size_t)B_ * H_ * S_ * DK_ * 2;  // 8 MB
  _Float16* ks = (_Float16*)ws; ws += (size_t)B_ * H_ * S_ * DK_ * 2;  // 8 MB
  _Float16* vs = (_Float16*)ws; ws += (size_t)B_ * S_ * DK_ * 2;       // 1 MB
  float* heads = (float*)ws;                                           // 16 MB

  proj_kernel<<<dim3(64, 17), 256, 0, stream>>>(q, k, v, Wq, bq, Wk, bk, Wv, bv,
                                                qs, ks, vs);
  attn_kernel<<<dim3(S_ / 16, B_ * H_), 256, 0, stream>>>(qs, ks, vs, out_attn,
                                                          heads);
  out_kernel<<<dim3((B_ * S_) / 128), 256, 0, stream>>>(heads, Wh, out_main);
}
